// SoftPointRenderer_73418170957844
// MI455X (gfx1250) — matrix operand order repaired and checked by reading, not compiled, not run on hardware
//
#include <hip/hip_runtime.h>
#include <hip/hip_bf16.h>

typedef __attribute__((ext_vector_type(16))) _Float16 v16h;
typedef __attribute__((ext_vector_type(8)))  float    v8f;

#define IMG_H 96
#define IMG_W 96
#define NB    4
#define NP    2048            // K*N = 8*256 points per batch
#define SIGMA 1.5f
#define ZMIN  0.2f
#define ZMAX  10.0f
#define FXY   60.0f

// ---------------------------------------------------------------------------
// Phase 1: one thread per (b, col, p) exp evaluation.
//   col in [0,96)   -> Ex weight for pixel x = col   (uses u)
//   col in [96,192) -> Ey weight for pixel y = col-96 (uses v, * valid)
// Output layouts (transposed, K-contiguous for WMMA fragment loads):
//   exT[b][w][p], eyT[b][h][p]   as _Float16, stride NP in p.
// ---------------------------------------------------------------------------
__global__ void splat_weights_f16(const float* __restrict__ pts,
                                  _Float16* __restrict__ exT,
                                  _Float16* __restrict__ eyT) {
    int tid  = blockIdx.x * blockDim.x + threadIdx.x;
    int p    = tid & (NP - 1);
    int rest = tid >> 11;                 // NP == 2^11
    int col  = rest % 192;
    int b    = rest / 192;
    if (b >= NB) return;

    const float* pw = pts + ((size_t)(b * NP + p)) * 3;
    float x = pw[0];
    float y = pw[1];
    float z = fmaxf(pw[2], 0.001f);

    const float cx = (IMG_W - 1) * 0.5f;
    const float cy = (IMG_H - 1) * 0.5f;
    const float inv_sigma = 1.0f / SIGMA;

    float e;
    if (col < 96) {
        float u = FXY * (x / z) + cx;
        float d = ((float)col - u) * inv_sigma;
        e = __expf(-0.5f * d * d);
        exT[((size_t)(b * IMG_W + col)) * NP + p] = (_Float16)e;
    } else {
        float v = FXY * (y / z) + cy;
        float d = ((float)(col - 96) - v) * inv_sigma;
        e = __expf(-0.5f * d * d);
        float valid = (z > ZMIN && z < ZMAX) ? 1.0f : 0.0f;
        eyT[((size_t)(b * IMG_H + (col - 96))) * NP + p] = (_Float16)(e * valid);
    }
}

// ---------------------------------------------------------------------------
// Phase 2: per-batch GEMM  img[b] = Ey^T (96xP) * Ex (Px96), P = 2048,
// tiled 16x16, one wave per tile, K stepped by 32 with
// V_WMMA_F32_16X16X32_F16 (f32 accumulate). Epilogue: 1 - exp(-acc).
//
// Fragment layouts (CDNA5 ISA 7.12.2, wave32):
//   A 16x32 f16 : lane l -> m = l&15, g = l>>4;
//                 VGPR i<4 : K = 2i   + 8g (+1 in hi half)
//                 VGPR i>=4: K = 16 + 2(i-4) + 8g
//   B 32x16 f16 : lane l -> n = l&15, g = l>>4;
//                 VGPR i   : K as for A (i<4: 2i + 8g, i>=4: 16 + 2(i-4) + 8g; +1 in hi half)
//   D 16x16 f32 : VGPR r -> row m = r + 8g, col n = l&15
// A[m,k] = eyT[b][m][k]  (K-contiguous -> one u32 load per VGPR)
// B[k,n] = exT[b][n][k]  (K-contiguous -> one u32 load per VGPR)
// ---------------------------------------------------------------------------
__global__ void splat_gemm_wmma(const _Float16* __restrict__ exT,
                                const _Float16* __restrict__ eyT,
                                float* __restrict__ out) {
    int wave = (blockIdx.x * blockDim.x + threadIdx.x) >> 5;   // 0..143
    int lane = threadIdx.x & 31;

    int b  = wave / 36;          // batch
    int t  = wave % 36;          // tile within 6x6 grid
    int th = t / 6;              // tile row  (h)
    int tw = t % 6;              // tile col  (w)

    int g  = lane >> 4;          // lane group (0: lanes 0-15, 1: lanes 16-31)
    int m  = (lane & 15) + th * 16;   // output row (h) this lane feeds for A
    int n  = (lane & 15) + tw * 16;   // output col (w) this lane feeds for B/D

    const _Float16* Arow = eyT + ((size_t)(b * IMG_H + m)) * NP;  // A[m, k]
    const _Float16* Brow = exT + ((size_t)(b * IMG_W + n)) * NP;  // B[k, n]

    v8f c = {};
    for (int k0 = 0; k0 < NP; k0 += 32) {
        v16h a, bm;
        unsigned int* ap = (unsigned int*)&a;
        unsigned int* bp = (unsigned int*)&bm;
#pragma unroll
        for (int i = 0; i < 8; ++i) {
            int ka = k0 + 8 * g + ((i < 4) ? (2 * i) : (16 + 2 * (i - 4)));
            int kb = k0 + 8 * g + ((i < 4) ? (2 * i) : (16 + 2 * (i - 4)));
            ap[i] = *(const unsigned int*)(Arow + ka);   // 2 consecutive-K f16
            bp[i] = *(const unsigned int*)(Brow + kb);
        }
        c = __builtin_amdgcn_wmma_f32_16x16x32_f16(
                /*neg_a=*/false, a, /*neg_b=*/false, bm,
                /*c_mod=*/(short)0, c, /*reuse_a=*/false, /*reuse_b=*/false);
    }

    // Epilogue: img = 1 - exp(-acc); D layout -> row = r + 8g (+ tile offset)
    float* obase = out + (size_t)b * IMG_H * IMG_W;
#pragma unroll
    for (int r = 0; r < 8; ++r) {
        int h = th * 16 + r + 8 * g;
        obase[(size_t)h * IMG_W + n] = 1.0f - __expf(-c[r]);
    }
}

// ---------------------------------------------------------------------------
extern "C" void kernel_launch(void* const* d_in, const int* in_sizes, int n_in,
                              void* d_out, int out_size, void* d_ws, size_t ws_size,
                              hipStream_t stream) {
    const float* pts = (const float*)d_in[0];        // (4,8,256,3) f32
    float* out       = (float*)d_out;                // (4,1,96,96) f32

    _Float16* exT = (_Float16*)d_ws;                 // (4,96,2048) f16
    _Float16* eyT = exT + (size_t)NB * IMG_W * NP;   // (4,96,2048) f16

    // Phase 1: 4 * 192 * 2048 = 1,572,864 threads -> 6144 blocks of 256
    splat_weights_f16<<<6144, 256, 0, stream>>>(pts, exT, eyT);

    // Phase 2: 144 tiles, 4 waves (128 threads) per block -> 36 blocks
    splat_gemm_wmma<<<36, 128, 0, stream>>>(exT, eyT, out);
}
